// RotaryAttention_86320252715347
// MI455X (gfx1250) — compile-verified, matrix operand order checked by reading, not run on hardware
//
#include <hip/hip_runtime.h>

#define HID   2048
#define NHEAD 16
#define HD    128
#define BATCH 4
#define SEQ   1024
#define MROWS (BATCH*SEQ)   // 4096

typedef __attribute__((ext_vector_type(16))) __bf16 v16bf;
typedef __attribute__((ext_vector_type(8)))  float  v8f;

union FragU { uint4 u[2]; v16bf v; __bf16 e[16]; };

// Load a 16x32 bf16 fragment slice for one lane: two contiguous 16B chunks
// (K = h8+0..7 and K = 16+h8+0..7 of one row), per the CDNA5 A/B VGPR layout.
static __device__ __forceinline__ v16bf ld_frag(const __bf16* p) {
    FragU r;
    r.u[0] = *(const uint4*)(p);
    r.u[1] = *(const uint4*)(p + 16);
    return r.v;
}

static __device__ __forceinline__ v8f wmma_bf16(v16bf a, v16bf b, v8f c) {
    // D = A(16x32) * B(32x16) + C, f32 accumulate
    return __builtin_amdgcn_wmma_f32_16x16x32_bf16(false, a, false, b,
                                                   (short)0, c, false, false);
}

static __device__ __forceinline__ v8f zero_v8f() {
    v8f z = {0.f,0.f,0.f,0.f,0.f,0.f,0.f,0.f};
    return z;
}

// ---- CDNA5 async copy: global -> LDS, 16B per lane, tracked by ASYNCcnt.
// LDS destination is the low 32 bits of the generic shared-memory address
static __device__ __forceinline__ void async_ld_b128(void* lds, const void* g) {
    unsigned off = (unsigned)(uintptr_t)lds;
    asm volatile("global_load_async_to_lds_b128 %0, %1, off"
                 :: "v"(off), "v"(g)
                 : "memory");
}
static __device__ __forceinline__ void wait_async() {
    asm volatile("s_wait_asynccnt 0x0" ::: "memory");
}

// ---------------------------------------------------------------- converts
__global__ void cvt_f32_bf16(const float* __restrict__ s, __bf16* __restrict__ d, size_t n) {
    size_t i = (size_t)blockIdx.x * blockDim.x + threadIdx.x;
    size_t st = (size_t)gridDim.x * blockDim.x;
    for (; i < n; i += st) d[i] = (__bf16)s[i];
}

__global__ void zero_f32(float* __restrict__ p, size_t n) {
    size_t i = (size_t)blockIdx.x * blockDim.x + threadIdx.x;
    size_t st = (size_t)gridDim.x * blockDim.x;
    for (; i < n; i += st) p[i] = 0.0f;
}

// ---------------------------------------------------------------- NT GEMM
// C[m,n] = sum_k A[m,k]*Bw[n,k] + bias[n]   (A: MROWSxHID, Bw: HIDxHID, bf16)
// mode 0: write f32 row-major to outF (final output projection)
// mode 1: *pe, *1/sqrt(HD), write bf16 (B,H,S,D)  (Q)
// mode 2: *pe,              write bf16 (B,H,S,D)  (K)
// mode 3:                   write bf16 (B,H,S,D)  (V)
__global__ __launch_bounds__(256)
void gemm_nt(const __bf16* __restrict__ A, const __bf16* __restrict__ Bw,
             const float* __restrict__ bias, const float* __restrict__ pe,
             float* __restrict__ outF, __bf16* __restrict__ outB, int mode)
{
    __shared__ __bf16 As[128 * 40];   // 128 rows x 32 K, stride 40 (16B aligned rows)
    __shared__ __bf16 Bs[128 * 40];

    const int tid  = threadIdx.x;
    const int lane = tid & 31;
    const int wave = tid >> 5;        // 8 waves
    const int lr   = lane & 15;
    const int h8   = (lane >> 4) << 3;
    const int wm   = (wave & 3) * 32; // 4 waves over M
    const int wn   = (wave >> 2) * 64;// 2 waves over N
    const int n0   = blockIdx.x * 128;
    const int m0   = blockIdx.y * 128;

    v8f acc[2][4];
    #pragma unroll
    for (int i = 0; i < 2; ++i)
        #pragma unroll
        for (int j = 0; j < 4; ++j) acc[i][j] = zero_v8f();

    for (int k = 0; k < HID; k += 32) {
        // stage 128x32 tiles of A and Bw: async global->LDS (ASYNCcnt path)
        #pragma unroll
        for (int ch = tid; ch < 512; ch += 256) {
            int row = ch >> 2;
            int cc  = (ch & 3) << 3;
            async_ld_b128(&As[row * 40 + cc],
                          &A[(size_t)(m0 + row) * HID + k + cc]);
            async_ld_b128(&Bs[row * 40 + cc],
                          &Bw[(size_t)(n0 + row) * HID + k + cc]);
        }
        wait_async();
        __syncthreads();

        v16bf af[2], bfrag[4];
        #pragma unroll
        for (int fm = 0; fm < 2; ++fm)
            af[fm] = ld_frag(&As[(wm + fm * 16 + lr) * 40 + h8]);
        #pragma unroll
        for (int fn = 0; fn < 4; ++fn)
            bfrag[fn] = ld_frag(&Bs[(wn + fn * 16 + lr) * 40 + h8]);
        #pragma unroll
        for (int fm = 0; fm < 2; ++fm)
            #pragma unroll
            for (int fn = 0; fn < 4; ++fn)
                acc[fm][fn] = wmma_bf16(af[fm], bfrag[fn], acc[fm][fn]);
        __syncthreads();
    }

    const float scaleQ = 0.08838834764831845f; // 1/sqrt(128)
    #pragma unroll
    for (int fm = 0; fm < 2; ++fm) {
        #pragma unroll
        for (int fn = 0; fn < 4; ++fn) {
            const int n  = n0 + wn + fn * 16 + lr;
            const float bv = bias[n];
            #pragma unroll
            for (int j = 0; j < 8; ++j) {
                const int m = m0 + wm + fm * 16 + j + h8;  // C: M = vgpr + (lane>=16)*8
                float v = acc[fm][fn][j] + bv;
                if (mode == 0) {
                    outF[(size_t)m * HID + n] = v;
                } else {
                    if (mode < 3)
                        v *= pe[(size_t)(m & (SEQ - 1)) * HD + (n & (HD - 1))];
                    if (mode == 1) v *= scaleQ;
                    // (B,H,S,D): b=m>>10, h=n>>7, s=m&1023, d=n&127
                    size_t idx = (((size_t)(m >> 10) * NHEAD + (n >> 7)) << 17)
                               + ((size_t)(m & (SEQ - 1)) << 7) + (n & (HD - 1));
                    outB[idx] = (__bf16)v;
                }
            }
        }
    }
}

// ---------------------------------------------------------------- attention
// One block = (b, h, 32-row q tile). Dynamic LDS: 32x1032 f32 scores + 32x136 bf16 Q.
__global__ __launch_bounds__(256)
void attn_kernel(const __bf16* __restrict__ Q, const __bf16* __restrict__ K,
                 const __bf16* __restrict__ V, const unsigned char* __restrict__ mask,
                 float* __restrict__ attn_mean, __bf16* __restrict__ attOut)
{
    extern __shared__ __align__(16) char smem_raw[];
    float*  Ssc = (float*)smem_raw;                                 // 32 x 1032
    __bf16* Qs  = (__bf16*)(smem_raw + 32 * 1032 * sizeof(float));  // 32 x 136

    const int tid  = threadIdx.x;
    const int lane = tid & 31;
    const int wave = tid >> 5;
    const int lr   = lane & 15;
    const int h8   = (lane >> 4) << 3;

    const int qt = blockIdx.x & 31;   // 32 q-tiles of 32 rows
    const int bh = blockIdx.x >> 5;   // 0..63
    const int b  = bh >> 4;
    const int h  = bh & 15;
    const size_t headBase = (size_t)bh * SEQ * HD;

    // Q tile 32x128 -> LDS via async copy
    for (int ch = tid; ch < 512; ch += 256) {
        int row = ch >> 4;
        int cc  = (ch & 15) << 3;
        async_ld_b128(&Qs[row * 136 + cc],
                      &Q[headBase + (size_t)(qt * 32 + row) * HD + cc]);
    }
    wait_async();
    __syncthreads();

    // ---- phase 1: scores = Q*K^T (scale pre-folded into Q). Wave -> 128 key cols.
    {
        v8f acc[2][8];
        #pragma unroll
        for (int i = 0; i < 2; ++i)
            #pragma unroll
            for (int j = 0; j < 8; ++j) acc[i][j] = zero_v8f();

        const int nc0 = wave * 128;
        #pragma unroll
        for (int kk = 0; kk < HD; kk += 32) {
            v16bf af[2];
            #pragma unroll
            for (int fm = 0; fm < 2; ++fm)
                af[fm] = ld_frag(&Qs[(fm * 16 + lr) * 136 + kk + h8]);
            #pragma unroll
            for (int fn = 0; fn < 8; ++fn) {
                v16bf bfrag = ld_frag(
                    &K[headBase + (size_t)(nc0 + fn * 16 + lr) * HD + kk + h8]);
                #pragma unroll
                for (int fm = 0; fm < 2; ++fm)
                    acc[fm][fn] = wmma_bf16(af[fm], bfrag, acc[fm][fn]);
            }
        }
        #pragma unroll
        for (int fn = 0; fn < 8; ++fn) {
            const int c  = nc0 + fn * 16 + lr;
            const bool ok = mask[(size_t)b * SEQ + c] != 0;
            #pragma unroll
            for (int fm = 0; fm < 2; ++fm)
                #pragma unroll
                for (int j = 0; j < 8; ++j) {
                    int r = fm * 16 + j + h8;
                    Ssc[r * 1032 + c] = ok ? acc[fm][fn][j] : -1.0e9f;
                }
        }
    }
    __syncthreads();

    // ---- phase 2: row softmax (8 lanes per row) + attn-mean atomics
    {
        const int sub = lane & 7;
        const int row = wave * 4 + (lane >> 3);
        float* Srow = &Ssc[row * 1032];
        float mx = -3.0e38f;
        for (int c = sub; c < SEQ; c += 8) mx = fmaxf(mx, Srow[c]);
        #pragma unroll
        for (int o = 1; o < 8; o <<= 1) mx = fmaxf(mx, __shfl_xor(mx, o, 32));
        float sum = 0.f;
        for (int c = sub; c < SEQ; c += 8) {
            float e = __expf(Srow[c] - mx);
            Srow[c] = e;
            sum += e;
        }
        #pragma unroll
        for (int o = 1; o < 8; o <<= 1) sum += __shfl_xor(sum, o, 32);
        const float inv = 1.0f / sum;
        const int qg = qt * 32 + row;
        float* mrow = attn_mean + ((size_t)b * SEQ + qg) * SEQ;
        for (int c = sub; c < SEQ; c += 8) {
            float p = Srow[c] * inv;
            Srow[c] = p;
            atomicAdd(&mrow[c], p * (1.0f / NHEAD));
        }
    }
    __syncthreads();

    // ---- phase 3: attended = P @ V. Wave -> 16 d-cols.
    {
        v8f acc[2] = { zero_v8f(), zero_v8f() };
        const int d0 = wave * 16;
        for (int k = 0; k < SEQ; k += 32) {
            FragU fb; // V columns (strided gather)
            #pragma unroll
            for (int j = 0; j < 8; ++j) {
                fb.e[j]     = V[headBase + (size_t)(k + h8 + j) * HD + d0 + lr];
                fb.e[j + 8] = V[headBase + (size_t)(k + 16 + h8 + j) * HD + d0 + lr];
            }
            #pragma unroll
            for (int fm = 0; fm < 2; ++fm) {
                FragU fa; // P rows, f32 LDS -> bf16
                const float* src = &Ssc[(fm * 16 + lr) * 1032 + k + h8];
                #pragma unroll
                for (int j = 0; j < 8; ++j) {
                    fa.e[j]     = (__bf16)src[j];
                    fa.e[j + 8] = (__bf16)src[16 + j];
                }
                acc[fm] = wmma_bf16(fa.v, fb.v, acc[fm]);
            }
        }
        #pragma unroll
        for (int fm = 0; fm < 2; ++fm)
            #pragma unroll
            for (int j = 0; j < 8; ++j) {
                int qg = qt * 32 + fm * 16 + j + h8;
                attOut[(((size_t)b * SEQ + qg) * NHEAD + h) * HD + d0 + lr] =
                    (__bf16)acc[fm][j];
            }
    }
}

// ---------------------------------------------------------------- launcher
extern "C" void kernel_launch(void* const* d_in, const int* in_sizes, int n_in,
                              void* d_out, int out_size, void* d_ws, size_t ws_size,
                              hipStream_t stream)
{
    (void)in_sizes; (void)n_in; (void)out_size; (void)ws_size;

    const float* q_in = (const float*)d_in[0];
    const float* k_in = (const float*)d_in[1];
    const float* v_in = (const float*)d_in[2];
    const unsigned char* mask = (const unsigned char*)d_in[3];
    const float* Wq = (const float*)d_in[4];
    const float* bq = (const float*)d_in[5];
    const float* Wk = (const float*)d_in[6];
    const float* bk = (const float*)d_in[7];
    const float* Wv = (const float*)d_in[8];
    const float* bv = (const float*)d_in[9];
    const float* Wo = (const float*)d_in[10];
    const float* bo = (const float*)d_in[11];
    const float* rot = (const float*)d_in[12];

    const size_t NX = (size_t)MROWS * HID;   // 8,388,608 activation elems
    const size_t NW = (size_t)HID * HID;     // 4,194,304 weight elems

    char* ws = (char*)d_ws;
    __bf16* xq  = (__bf16*)ws;               ws += NX * 2;
    __bf16* xk  = (__bf16*)ws;               ws += NX * 2;
    __bf16* xv  = (__bf16*)ws;               ws += NX * 2;
    __bf16* wqb = (__bf16*)ws;               ws += NW * 2;
    __bf16* wkb = (__bf16*)ws;               ws += NW * 2;
    __bf16* wvb = (__bf16*)ws;               ws += NW * 2;
    __bf16* wob = (__bf16*)ws;               ws += NW * 2;
    __bf16* Qb  = (__bf16*)ws;               ws += NX * 2;
    __bf16* Kb  = (__bf16*)ws;               ws += NX * 2;
    __bf16* Vb  = (__bf16*)ws;               ws += NX * 2;
    __bf16* Ab  = (__bf16*)ws;               ws += NX * 2;   // attended (B,S,H*D)

    float* outMain = (float*)d_out;                          // (B,S,HID)
    float* outMean = outMain + (size_t)MROWS * HID;          // (B,S,S)

    // f32 -> bf16
    cvt_f32_bf16<<<2048, 256, 0, stream>>>(q_in, xq, NX);
    cvt_f32_bf16<<<2048, 256, 0, stream>>>(k_in, xk, NX);
    cvt_f32_bf16<<<2048, 256, 0, stream>>>(v_in, xv, NX);
    cvt_f32_bf16<<<1024, 256, 0, stream>>>(Wq, wqb, NW);
    cvt_f32_bf16<<<1024, 256, 0, stream>>>(Wk, wkb, NW);
    cvt_f32_bf16<<<1024, 256, 0, stream>>>(Wv, wvb, NW);
    cvt_f32_bf16<<<1024, 256, 0, stream>>>(Wo, wob, NW);

    // zero attn-mean accumulation target
    zero_f32<<<1024, 256, 0, stream>>>(outMean, (size_t)BATCH * SEQ * SEQ);

    // projections (NT GEMM + bias + pe/scale, write (B,H,S,D) bf16)
    dim3 gGemm(HID / 128, MROWS / 128);
    gemm_nt<<<gGemm, 256, 0, stream>>>(xq, wqb, bq, rot, nullptr, Qb, 1);
    gemm_nt<<<gGemm, 256, 0, stream>>>(xk, wkb, bk, rot, nullptr, Kb, 2);
    gemm_nt<<<gGemm, 256, 0, stream>>>(xv, wvb, bv, nullptr, nullptr, Vb, 3);

    // attention: 4*16*32 = 2048 blocks, ~138 KB dynamic LDS per workgroup
    size_t smem = 32 * 1032 * sizeof(float) + 32 * 136 * sizeof(__bf16);
    attn_kernel<<<BATCH * NHEAD * (SEQ / 32), 256, smem, stream>>>(
        Qb, Kb, Vb, mask, outMean, Ab);

    // output projection -> f32 d_out
    gemm_nt<<<gGemm, 256, 0, stream>>>(Ab, wob, bo, nullptr, outMain, nullptr, 0);
}
